// TextGuidedPruning_73177652789418
// MI455X (gfx1250) — matrix operand order repaired and checked by reading, not compiled, not run on hardware
//
#include <hip/hip_runtime.h>

typedef float    v8f  __attribute__((ext_vector_type(8)));
typedef _Float16 v16h __attribute__((ext_vector_type(16)));
typedef _Float16 v8h  __attribute__((ext_vector_type(8)));
typedef _Float16 v4h  __attribute__((ext_vector_type(4)));

#define CC 256
#define BB 16
#define LLEN 32
#define HH 8
#define MT 32            // rows per block tile
#define SD 260           // f32 LDS row stride (banks: 260%64==4)
#define SDH 264          // f16 LDS row stride in halves (16B-aligned frag reads)
#define RSQRT_DH 0.17677669529663687f

// ---------------------------------------------------------------------------
__global__ void k_init(float* var_sum, unsigned* sim_mm, unsigned* imp_mm) {
    int t = threadIdx.x;
    if (t < BB) {
        var_sum[t] = 0.f;
        sim_mm[t * 2]     = 0x7f800000u;
        sim_mm[t * 2 + 1] = 0u;
        imp_mm[t * 2]     = 0x7f800000u;
        imp_mm[t * 2 + 1] = 0u;
    }
}

// ---------------------------------------------------------------------------
// Wqc[c][j] = sum_i Wq[c][i] * vp_w[i][j];  bqc[c] = Wq[c]·vp_b + bq[c]
__global__ void k_wqc(const float* __restrict__ ipw, const float* __restrict__ ipb,
                      const float* __restrict__ vp_w, const float* __restrict__ vp_b,
                      float* __restrict__ Wqc, float* __restrict__ bqc) {
    int c = blockIdx.x, j = threadIdx.x;
    const float* wq = ipw + (size_t)c * CC;
    float acc = 0.f;
    for (int i = 0; i < CC; ++i) acc += wq[i] * vp_w[(size_t)i * CC + j];
    Wqc[(size_t)c * CC + j] = acc;
    if (j == 0) {
        float a = 0.f;
        for (int i = 0; i < CC; ++i) a += wq[i] * vp_b[i];
        bqc[c] = a + ipb[c];
    }
}

// ---------------------------------------------------------------------------
// pack w1 (256 out-cols) and w2 (128 out-cols) into f16 WMMA-B fragment order.
// B[k][col] = w[col*256+k].  Fragment chunk = (tile,kb): 512 halves, lane-major:
//   chunk[(kh*16+c)*16 + j] = B[kb*32 + kh*16 + j][tile*16 + c]
__global__ void k_pack(const float* __restrict__ w1, const float* __restrict__ w2,
                       _Float16* __restrict__ w1pk, _Float16* __restrict__ w2pk) {
    int idx = blockIdx.x * 256 + threadIdx.x;           // 65536 threads
    {
        int k = idx >> 8, col = idx & 255;
        int tile = col >> 4, c = col & 15, kb = k >> 5, kh = (k >> 4) & 1, j = k & 15;
        w1pk[((size_t)(tile * 8 + kb)) * 512 + (kh * 16 + c) * 16 + j] =
            (_Float16)w1[(size_t)col * CC + k];
    }
    if (idx < CC * 128) {
        int k = idx >> 7, col = idx & 127;
        int tile = col >> 4, c = col & 15, kb = k >> 5, kh = (k >> 4) & 1, j = k & 15;
        w2pk[((size_t)(tile * 8 + kb)) * 512 + (kh * 16 + c) * 16 + j] =
            (_Float16)w2[(size_t)col * CC + k];
    }
}

// ---------------------------------------------------------------------------
// text_proj[lb][i] = tp_b[i] + T[lb]·tp_w[i]
__global__ void k_tproj(const float* __restrict__ T, const float* __restrict__ tp_w,
                        const float* __restrict__ tp_b, float* __restrict__ tproj) {
    __shared__ float row[CC];
    int lb = blockIdx.x, i = threadIdx.x;
    row[i] = T[(size_t)lb * CC + i];
    __syncthreads();
    const float* w = tp_w + (size_t)i * CC;
    float acc = tp_b[i];
    for (int j = 0; j < CC; ++j) acc += row[j] * w[j];
    tproj[(size_t)lb * CC + i] = acc;
}

// kp[lb][c] = bk[c] + tproj[lb]·Wk[c]
__global__ void k_kp(const float* __restrict__ tproj, const float* __restrict__ ipw,
                     const float* __restrict__ ipb, float* __restrict__ kp) {
    __shared__ float row[CC];
    int lb = blockIdx.x, c = threadIdx.x;
    row[c] = tproj[(size_t)lb * CC + c];
    __syncthreads();
    const float* wk = ipw + (size_t)(CC + c) * CC;
    float acc = ipb[CC + c];
    for (int i = 0; i < CC; ++i) acc += row[i] * wk[i];
    kp[(size_t)lb * CC + c] = acc;
}

// ---------------------------------------------------------------------------
// P[b][k][col=h*32+l] = (1/sqrt(DH)) * sum_d Wqc[h*32+d][k]*kp[l,b,h*32+d]
// written directly in f16 fragment-packed order; s0[b][col] = bias row (f32)
__global__ void k_makeP(const float* __restrict__ Wqc, const float* __restrict__ bqc,
                        const float* __restrict__ kp, _Float16* __restrict__ Ppk,
                        float* __restrict__ s0) {
    int k = blockIdx.x, b = blockIdx.y, col = threadIdx.x;
    int h = col >> 5, l = col & 31;
    const float* kr = kp + ((size_t)(l * BB + b)) * CC + h * 32;
    float acc = 0.f;
    for (int d = 0; d < 32; ++d) acc += Wqc[(size_t)(h * 32 + d) * CC + k] * kr[d];
    acc *= RSQRT_DH;
    int tile = col >> 4, c = col & 15, kb = k >> 5, kh = (k >> 4) & 1, j = k & 15;
    Ppk[((size_t)b << 16) + ((size_t)(tile * 8 + kb)) * 512 + (kh * 16 + c) * 16 + j] =
        (_Float16)acc;
    if (k == 0) {
        float a = 0.f;
        for (int d = 0; d < 32; ++d) a += bqc[h * 32 + d] * kr[d];
        s0[b * CC + col] = a * RSQRT_DH;
    }
}

// ---------------------------------------------------------------------------
__device__ __forceinline__ v16h load_afrag(const _Float16* __restrict__ arow, int kb, int h) {
    const _Float16* p = arow + kb * 32 + 16 * h;
    v8h lo = *(const v8h*)p;          // K = kb + 16h .. +7     -> elems 0..7
    v8h hi = *(const v8h*)(p + 8);    // K = kb + 16h + 8 .. +7 -> elems 8..15
    return __builtin_shufflevector(lo, hi, 0, 1, 2, 3, 4, 5, 6, 7,
                                   8, 9, 10, 11, 12, 13, 14, 15);
}

// ---------------------------------------------------------------------------
// Fused per (b, 32-row n-tile):
//   scores = X @ P[b] + s0   (f16 WMMA, f32 acc) -> softmax/head-mean/var/top16
//   h1 = relu(LN(X @ w1 + b1)); h2 = relu(LN(h1 @ w2 + b2)); imp = sigm(h2·w3+b3)
// Wave assignment: each wave owns 32 output cols and BOTH 16-row tiles, so every
// B fragment is loaded once per block and reused by two WMMAs.
__global__ __launch_bounds__(256) void k_main(
    const float* __restrict__ X, const _Float16* __restrict__ Ppk, const float* __restrict__ s0,
    const _Float16* __restrict__ w1pk, const float* __restrict__ b1,
    const float* __restrict__ g1v, const float* __restrict__ be1v,
    const _Float16* __restrict__ w2pk, const float* __restrict__ b2,
    const float* __restrict__ g2v, const float* __restrict__ be2v,
    const float* __restrict__ w3, const float* __restrict__ b3,
    float* __restrict__ top16, float* __restrict__ impb,
    float* __restrict__ var_sum, unsigned* __restrict__ imp_mm, int Ntot) {
    __shared__ _Float16 ls_x[MT * SDH];   // X tile (f16), then h1 (f16)
    __shared__ float    ls_s[MT * SD];    // scores / h1(f32) / h2(f32)
    __shared__ float    ls_p[896];        // g1 be1 g2 be2 w3
    __shared__ float    ls_red[MT];

    const int tid   = threadIdx.x;
    const int lane  = tid & 31;
    const int wv    = tid >> 5;
    const int bb    = blockIdx.y;
    const int n0    = blockIdx.x * MT;
    const int rr    = lane & 15;
    const int hh2   = lane >> 4;
    const int rbase = hh2 << 3;

    for (int i = tid; i < 896; i += 256) {
        float v;
        if (i < 256)      v = g1v[i];
        else if (i < 512) v = be1v[i - 256];
        else if (i < 640) v = g2v[i - 512];
        else if (i < 768) v = be2v[i - 640];
        else              v = w3[i - 768];
        ls_p[i] = v;
    }
    // stage X tile (32 rows x 256), f32 -> f16
    for (int idx = tid; idx < (MT * CC) / 4; idx += 256) {
        int e = idx * 4, r = e >> 8, c = e & 255;
        float4 vv = *(const float4*)(X + (((size_t)(n0 + r) * BB + bb) << 8) + c);
        v4h hh = {(_Float16)vv.x, (_Float16)vv.y, (_Float16)vv.z, (_Float16)vv.w};
        *(v4h*)&ls_x[r * SDH + c] = hh;
    }
    __syncthreads();

    const _Float16* arow0 = &ls_x[rr * SDH];          // row tile 0
    const _Float16* arow1 = &ls_x[(16 + rr) * SDH];   // row tile 1

    // ---- GEMM 1: attention scores (8 waves x 2 col tiles x 2 row tiles) ----
    {
        const int c0 = wv * 32;
        v8f a00, a01, a10, a11;               // [rowtile][coltile]
        {
            float sv0 = s0[bb * CC + c0 + rr];
            float sv1 = s0[bb * CC + c0 + 16 + rr];
            for (int i = 0; i < 8; ++i) { a00[i] = sv0; a10[i] = sv0; a01[i] = sv1; a11[i] = sv1; }
        }
        const _Float16* Bp = Ppk + ((size_t)bb << 16);
#pragma unroll
        for (int kb = 0; kb < 8; ++kb) {
            v16h x0 = load_afrag(arow0, kb, hh2);
            v16h x1 = load_afrag(arow1, kb, hh2);
            v16h bf0 = *(const v16h*)(Bp + ((size_t)((wv * 2 + 0) * 8 + kb)) * 512 + lane * 16);
            v16h bf1 = *(const v16h*)(Bp + ((size_t)((wv * 2 + 1) * 8 + kb)) * 512 + lane * 16);
            a00 = __builtin_amdgcn_wmma_f32_16x16x32_f16(false, x0, false, bf0, (short)0, a00, false, false);
            a10 = __builtin_amdgcn_wmma_f32_16x16x32_f16(false, x1, false, bf0, (short)0, a10, false, false);
            a01 = __builtin_amdgcn_wmma_f32_16x16x32_f16(false, x0, false, bf1, (short)0, a01, false, false);
            a11 = __builtin_amdgcn_wmma_f32_16x16x32_f16(false, x1, false, bf1, (short)0, a11, false, false);
        }
#pragma unroll
        for (int i = 0; i < 8; ++i) {
            ls_s[(rbase + i) * SD + c0 + rr]            = a00[i];
            ls_s[(16 + rbase + i) * SD + c0 + rr]       = a10[i];
            ls_s[(rbase + i) * SD + c0 + 16 + rr]       = a01[i];
            ls_s[(16 + rbase + i) * SD + c0 + 16 + rr]  = a11[i];
        }
    }
    __syncthreads();

    // ---- per-row: softmax/head -> mean attn[32], variance, top-16 ----
    if (tid < MT) {
        const int row = tid;
        float attn[LLEN];
        for (int l = 0; l < LLEN; ++l) attn[l] = 0.f;
        for (int h = 0; h < HH; ++h) {
            const float* sr = &ls_s[row * SD + h * LLEN];
            float mx = sr[0];
            for (int l = 1; l < LLEN; ++l) mx = fmaxf(mx, sr[l]);
            float z = 0.f, e[LLEN];
            for (int l = 0; l < LLEN; ++l) { e[l] = __expf(sr[l] - mx); z += e[l]; }
            float iz = 1.f / z;
            for (int l = 0; l < LLEN; ++l) attn[l] += e[l] * iz;
        }
        float mu = 0.f;
        for (int l = 0; l < LLEN; ++l) { attn[l] *= (1.f / HH); mu += attn[l]; }
        mu *= (1.f / LLEN);
        float var = 0.f;
        for (int l = 0; l < LLEN; ++l) { float d = attn[l] - mu; var += d * d; }
        ls_red[tid] = var * (1.f / (LLEN - 1));
        for (int i = 0; i < 16; ++i) {      // selection top-16 desc
            int mj = i; float mv = attn[i];
            for (int j = i + 1; j < LLEN; ++j) if (attn[j] > mv) { mv = attn[j]; mj = j; }
            float t = attn[i]; attn[i] = attn[mj]; attn[mj] = t;
        }
        float* tp = top16 + ((size_t)bb * Ntot + (n0 + row)) * 16;
        for (int i = 0; i < 16; ++i) tp[i] = attn[i];
    }
    __syncthreads();
    if (tid == 0) {
        float s = 0.f;
        for (int i = 0; i < MT; ++i) s += ls_red[i];
        atomicAdd(&var_sum[bb], s);
    }

    // ---- GEMM 2: h1 = X @ w1 + b1 (f32 into ls_s) ----
    {
        const int c0 = wv * 32;
        v8f a00, a01, a10, a11;
        {
            float sv0 = b1[c0 + rr];
            float sv1 = b1[c0 + 16 + rr];
            for (int i = 0; i < 8; ++i) { a00[i] = sv0; a10[i] = sv0; a01[i] = sv1; a11[i] = sv1; }
        }
#pragma unroll
        for (int kb = 0; kb < 8; ++kb) {
            v16h x0 = load_afrag(arow0, kb, hh2);
            v16h x1 = load_afrag(arow1, kb, hh2);
            v16h bf0 = *(const v16h*)(w1pk + ((size_t)((wv * 2 + 0) * 8 + kb)) * 512 + lane * 16);
            v16h bf1 = *(const v16h*)(w1pk + ((size_t)((wv * 2 + 1) * 8 + kb)) * 512 + lane * 16);
            a00 = __builtin_amdgcn_wmma_f32_16x16x32_f16(false, x0, false, bf0, (short)0, a00, false, false);
            a10 = __builtin_amdgcn_wmma_f32_16x16x32_f16(false, x1, false, bf0, (short)0, a10, false, false);
            a01 = __builtin_amdgcn_wmma_f32_16x16x32_f16(false, x0, false, bf1, (short)0, a01, false, false);
            a11 = __builtin_amdgcn_wmma_f32_16x16x32_f16(false, x1, false, bf1, (short)0, a11, false, false);
        }
#pragma unroll
        for (int i = 0; i < 8; ++i) {
            ls_s[(rbase + i) * SD + c0 + rr]            = a00[i];
            ls_s[(16 + rbase + i) * SD + c0 + rr]       = a10[i];
            ls_s[(rbase + i) * SD + c0 + 16 + rr]       = a01[i];
            ls_s[(16 + rbase + i) * SD + c0 + 16 + rr]  = a11[i];
        }
    }
    __syncthreads();

    // ---- LN1 + relu: f32 row in ls_s -> f16 row in ls_x ----
    if (tid < MT) {
        const float* r = &ls_s[tid * SD];
        _Float16* o = &ls_x[tid * SDH];
        float m = 0.f;
        for (int c = 0; c < CC; ++c) m += r[c];
        m *= (1.f / CC);
        float v = 0.f;
        for (int c = 0; c < CC; ++c) { float d = r[c] - m; v += d * d; }
        v *= (1.f / CC);
        float is = rsqrtf(v + 1e-5f);
        for (int c = 0; c < CC; ++c) {
            float y = (r[c] - m) * is * ls_p[c] + ls_p[256 + c];
            o[c] = (_Float16)fmaxf(y, 0.f);
        }
    }
    __syncthreads();

    // ---- GEMM 3: h2 = h1 @ w2 + b2 (8 waves x 1 col tile x 2 row tiles) ----
    {
        const int c0 = wv * 16;
        v8f a0, a1;
        {
            float sv = b2[c0 + rr];
            for (int i = 0; i < 8; ++i) { a0[i] = sv; a1[i] = sv; }
        }
#pragma unroll
        for (int kb = 0; kb < 8; ++kb) {
            v16h x0 = load_afrag(arow0, kb, hh2);
            v16h x1 = load_afrag(arow1, kb, hh2);
            v16h bf = *(const v16h*)(w2pk + ((size_t)(wv * 8 + kb)) * 512 + lane * 16);
            a0 = __builtin_amdgcn_wmma_f32_16x16x32_f16(false, x0, false, bf, (short)0, a0, false, false);
            a1 = __builtin_amdgcn_wmma_f32_16x16x32_f16(false, x1, false, bf, (short)0, a1, false, false);
        }
#pragma unroll
        for (int i = 0; i < 8; ++i) {
            ls_s[(rbase + i) * SD + c0 + rr]      = a0[i];
            ls_s[(16 + rbase + i) * SD + c0 + rr] = a1[i];
        }
    }
    __syncthreads();

    // ---- LN2 + relu + dot(w3) + sigmoid -> importance ----
    if (tid < MT) {
        const float* r = &ls_s[tid * SD];
        float m = 0.f;
        for (int c = 0; c < 128; ++c) m += r[c];
        m *= (1.f / 128.f);
        float v = 0.f;
        for (int c = 0; c < 128; ++c) { float d = r[c] - m; v += d * d; }
        v *= (1.f / 128.f);
        float is = rsqrtf(v + 1e-5f);
        float acc = 0.f;
        for (int c = 0; c < 128; ++c) {
            float y = (r[c] - m) * is * ls_p[512 + c] + ls_p[640 + c];
            acc += fmaxf(y, 0.f) * ls_p[768 + c];
        }
        acc += b3[0];
        float iv = 1.f / (1.f + __expf(-acc));
        impb[(size_t)bb * Ntot + n0 + tid] = iv;
        ls_red[tid] = iv;
    }
    __syncthreads();
    if (tid == 0) {
        float lo = ls_red[0], hi = ls_red[0];
        for (int i = 1; i < MT; ++i) { lo = fminf(lo, ls_red[i]); hi = fmaxf(hi, ls_red[i]); }
        atomicMin(&imp_mm[bb * 2],     __float_as_uint(lo));
        atomicMax(&imp_mm[bb * 2 + 1], __float_as_uint(hi));
    }
}

// ---------------------------------------------------------------------------
__global__ void k_ktop(const float* __restrict__ var_sum, const float* __restrict__ kw,
                       int* __restrict__ ktop, int Ntot) {
    int b = threadIdx.x;
    if (b >= BB) return;
    float var = var_sum[b] / (float)Ntot;
    float kws = 1.f / (1.f + expf(-kw[0]));
    float ratio = fminf(fmaxf(kws * (1.f + var * 0.5f), 0.25f), 0.6f);
    int kbase = (int)floorf((float)LLEN * ratio);
    if (kbase < 1) kbase = 1;
    int kt = kbase;
    if (kt > 16) kt = 16;
    if (kt > LLEN) kt = LLEN;
    if (kt < 8) kt = 8;
    ktop[b] = kt;
}

// ---------------------------------------------------------------------------
__global__ void k_sim(const float* __restrict__ top16, const int* __restrict__ ktop,
                      float* __restrict__ simb, unsigned* __restrict__ sim_mm, int Ntot) {
    __shared__ float red[256];
    int bb = blockIdx.y;
    int n = blockIdx.x * 256 + threadIdx.x;
    const float* tp = top16 + ((size_t)bb * Ntot + n) * 16;
    int K = ktop[bb];
    float t[16];
    for (int i = 0; i < 16; ++i) t[i] = tp[i];
    float mx = t[0];
    for (int i = 1; i < K; ++i) mx = fmaxf(mx, t[i]);
    float z = 0.f, s = 0.f;
    for (int i = 0; i < K; ++i) { float e = __expf(t[i] - mx); z += e; s += t[i] * e; }
    float sim = s / z;
    simb[(size_t)bb * Ntot + n] = sim;

    red[threadIdx.x] = sim;
    __syncthreads();
    for (int off = 128; off > 0; off >>= 1) {
        if (threadIdx.x < off) red[threadIdx.x] = fminf(red[threadIdx.x], red[threadIdx.x + off]);
        __syncthreads();
    }
    float lo = red[0];
    __syncthreads();
    red[threadIdx.x] = sim;
    __syncthreads();
    for (int off = 128; off > 0; off >>= 1) {
        if (threadIdx.x < off) red[threadIdx.x] = fmaxf(red[threadIdx.x], red[threadIdx.x + off]);
        __syncthreads();
    }
    if (threadIdx.x == 0) {
        atomicMin(&sim_mm[bb * 2],     __float_as_uint(lo));
        atomicMax(&sim_mm[bb * 2 + 1], __float_as_uint(red[0]));
    }
}

// ---------------------------------------------------------------------------
__global__ void k_final(const float* __restrict__ simb, const float* __restrict__ impb,
                        const unsigned* __restrict__ sim_mm, const unsigned* __restrict__ imp_mm,
                        const float* __restrict__ ac, const float* __restrict__ af,
                        float* __restrict__ out, int Ntot) {
    int idx = blockIdx.x * 256 + threadIdx.x;   // idx = n*B + b
    int n = idx >> 4, b = idx & 15;
    float alpha = 0.5f * (1.f / (1.f + expf(-ac[0])) + 1.f / (1.f + expf(-af[0])));
    float s = simb[(size_t)b * Ntot + n];
    float slo = __uint_as_float(sim_mm[b * 2]), shi = __uint_as_float(sim_mm[b * 2 + 1]);
    float srng = shi - slo;
    float sn = (srng > 0.f) ? (s - slo) / srng : 0.f;
    float p = impb[(size_t)b * Ntot + n];
    float plo = __uint_as_float(imp_mm[b * 2]), phi = __uint_as_float(imp_mm[b * 2 + 1]);
    float prng = phi - plo;
    float pn = (prng > 0.f) ? (p - plo) / prng : 0.f;
    out[idx] = alpha * sn + (1.f - alpha) * pn;
}

// ---------------------------------------------------------------------------
extern "C" void kernel_launch(void* const* d_in, const int* in_sizes, int n_in,
                              void* d_out, int out_size, void* d_ws, size_t ws_size,
                              hipStream_t stream) {
    const float* X    = (const float*)d_in[0];
    const float* T    = (const float*)d_in[1];
    const float* vp_w = (const float*)d_in[2];
    const float* vp_b = (const float*)d_in[3];
    const float* tp_w = (const float*)d_in[4];
    const float* tp_b = (const float*)d_in[5];
    const float* ipw  = (const float*)d_in[6];
    const float* ipb  = (const float*)d_in[7];
    const float* w1   = (const float*)d_in[8];
    const float* b1   = (const float*)d_in[9];
    const float* g1   = (const float*)d_in[10];
    const float* be1  = (const float*)d_in[11];
    const float* w2   = (const float*)d_in[12];
    const float* b2   = (const float*)d_in[13];
    const float* g2   = (const float*)d_in[14];
    const float* be2  = (const float*)d_in[15];
    const float* w3   = (const float*)d_in[16];
    const float* b3   = (const float*)d_in[17];
    const float* kw   = (const float*)d_in[18];
    const float* ac   = (const float*)d_in[19];
    const float* af   = (const float*)d_in[20];
    float* out = (float*)d_out;

    const int Ntot = in_sizes[0] / (BB * CC);   // 16384

    // workspace layout (float units; f16 buffers carved as aligned sub-ranges)
    float* ws = (float*)d_ws;
    size_t o = 0;
    float*     Wqc   = ws + o;             o += (size_t)CC * CC;        // 65536
    float*     bqc   = ws + o;             o += CC;
    float*     tproj = ws + o;             o += (size_t)LLEN * BB * CC; // 131072
    float*     kp    = ws + o;             o += (size_t)LLEN * BB * CC; // 131072
    _Float16*  Ppk   = (_Float16*)(ws + o); o += (size_t)BB * CC * CC / 2;  // 524288 fl
    float*     s0    = ws + o;             o += (size_t)BB * CC;
    _Float16*  w1pk  = (_Float16*)(ws + o); o += (size_t)CC * CC / 2;   // 32768 fl
    _Float16*  w2pk  = (_Float16*)(ws + o); o += (size_t)CC * 128 / 2;  // 16384 fl
    float*     top16 = ws + o;             o += (size_t)BB * Ntot * 16; // 4194304
    float*     simb  = ws + o;             o += (size_t)BB * Ntot;
    float*     impb  = ws + o;             o += (size_t)BB * Ntot;
    float*     var_s = ws + o;             o += BB;
    int*       ktop   = (int*)(ws + o);      o += BB;
    unsigned*  sim_mm = (unsigned*)(ws + o); o += 2 * BB;
    unsigned*  imp_mm = (unsigned*)(ws + o); o += 2 * BB;

    k_init<<<1, 64, 0, stream>>>(var_s, sim_mm, imp_mm);
    k_wqc<<<CC, CC, 0, stream>>>(ipw, ipb, vp_w, vp_b, Wqc, bqc);
    k_pack<<<256, 256, 0, stream>>>(w1, w2, w1pk, w2pk);
    k_tproj<<<LLEN * BB, CC, 0, stream>>>(T, tp_w, tp_b, tproj);
    k_kp<<<LLEN * BB, CC, 0, stream>>>(tproj, ipw, ipb, kp);
    k_makeP<<<dim3(CC, BB), CC, 0, stream>>>(Wqc, bqc, kp, Ppk, s0);
    k_main<<<dim3(Ntot / MT, BB), 256, 0, stream>>>(X, Ppk, s0, w1pk, b1, g1, be1,
                                                    w2pk, b2, g2, be2, w3, b3,
                                                    top16, impb, var_s, imp_mm, Ntot);
    k_ktop<<<1, 16, 0, stream>>>(var_s, kw, ktop, Ntot);
    k_sim<<<dim3(Ntot / 256, BB), 256, 0, stream>>>(top16, ktop, simb, sim_mm, Ntot);
    k_final<<<(Ntot * BB) / 256, 256, 0, stream>>>(simb, impb, sim_mm, imp_mm, ac, af, out, Ntot);
}
